// ToxicityLSTM_15857019257111
// MI455X (gfx1250) — matrix operand order repaired and checked by reading, not compiled, not run on hardware
//
#include <hip/hip_runtime.h>
#include <hip/hip_bf16.h>

// ============================================================================
// ToxicityLSTM on MI455X (gfx1250), compile-only tuned.
//
//  1. prep_b      : W0,U0,W1,U1 f32 -> f16 WMMA B-fragment layout.
//  2. embed_swz   : emb gather * padmask * in_mask0 * 2 -> f16 A-fragments.
//  3. gemm_xw     : XW = Xsw @ Wsw + b via v_wmma_f32_16x16x32_f16,
//                   output stored in D-fragment layout ([tile][lane][8 f16])
//                   so both producer and consumer use single b128 accesses.
//  4. lstm_rec    : persistent 1024-thread workgroup; h in LDS (A-fragment
//                   layout), c in VGPRs, U streamed from L2; next step's XW
//                   slab (128 KB, contiguous) double-buffered into LDS via
//                   GLOBAL_LOAD_ASYNC_TO_LDS_B128 (s_wait_asynccnt), hidden
//                   under the 64-WMMA GEMM phase.  Fallback: direct b128.
//  5-7. layer 1 (same machinery), 8. head.
// ============================================================================

typedef __attribute__((ext_vector_type(16))) _Float16 v16h;
typedef __attribute__((ext_vector_type(8)))  _Float16 v8h;
typedef __attribute__((ext_vector_type(8)))  float    v8f;
typedef __attribute__((ext_vector_type(4)))  int      v4i;

#define T_STEPS 256
#define BATCH   64
#define HID     256
#define GATES   1024              // 4*H
#define MROWS   (T_STEPS * BATCH) // 16384
#define STEP_HALVES (BATCH * GATES)      // 65536 f16 = 128 KB per time step

#if __has_builtin(__builtin_amdgcn_global_load_async_to_lds_b128)
#define HAVE_ASYNC_LDS 1
// Probe-derived signature: (int4 AS1*, int4 AS3*, imm offset, imm cpol)
typedef __attribute__((address_space(1))) v4i gv4i_t;
typedef __attribute__((address_space(3))) v4i sv4i_t;
#else
#define HAVE_ASYNC_LDS 0
#endif

__device__ __forceinline__ v8f wmma_f16(v16h a, v16h b, v8f c) {
    return __builtin_amdgcn_wmma_f32_16x16x32_f16(
        false, a, false, b, (short)0, c, false, false);
}

__device__ __forceinline__ float sigmoidf_(float x) {
    return 1.0f / (1.0f + __expf(-x));
}

// ---------------------------------------------------------------------------
// 1) [256,1024] f32 (K x N) -> f16 WMMA B-fragment layout
//    [ntile][kblk][lane][8 dwords]; lane = n%16 + 16*gB, dword v holds
//    k = gB*8 + (v&4?16:0) + (v&3)*2 + {0,1} (same as A-fragment layout).
// ---------------------------------------------------------------------------
__global__ __launch_bounds__(256) void prep_b_kernel(const float* __restrict__ src,
                                                     _Float16* __restrict__ dst) {
    int id   = blockIdx.x * 256 + threadIdx.x;   // 0 .. 131071
    int v    = id & 7;
    int lane = (id >> 3) & 31;
    int kblk = (id >> 8) & 7;
    int ntile = id >> 11;
    int laneN = lane & 15;
    int gB    = lane >> 4;
    int n  = ntile * 16 + laneN;
    int k0 = kblk * 32 + gB * 8 + ((v & 4) ? 16 : 0) + (v & 3) * 2;
    dst[id * 2 + 0] = (_Float16)src[(k0 + 0) * GATES + n];
    dst[id * 2 + 1] = (_Float16)src[(k0 + 1) * GATES + n];
}

// ---------------------------------------------------------------------------
// A-fragment layout (16x32 f16): [mtile][kblk][lane][8 dwords];
// lane = row%16 + 16*gA; dword v, half j -> K = gA*8 + (v&4?16:0) + (v&3)*2+j
// rows = t*B + b, K = feature column.
// ---------------------------------------------------------------------------
__global__ __launch_bounds__(256) void embed_swz_kernel(const int* __restrict__ tokens,
                                                        const float* __restrict__ emb,
                                                        const float* __restrict__ imask,
                                                        _Float16* __restrict__ dst) {
    int id    = blockIdx.x * 256 + threadIdx.x;  // 0 .. 2,097,151
    int v     = id & 7;
    int lane  = (id >> 3) & 31;
    int kblk  = (id >> 8) & 7;
    int mtile = id >> 11;
    int laneN = lane & 15;
    int gA    = lane >> 4;
    int row = mtile * 16 + laneN;   // = t*64 + b
    int t   = row >> 6;
    int b   = row & 63;
    int tok = tokens[b * T_STEPS + t];
    float scale = (tok != 1) ? 2.0f : 0.0f;   // pad mask fused with 1/KEEP_PROB
#pragma unroll
    for (int j = 0; j < 2; ++j) {
        int K   = gA * 8 + ((v & 4) ? 16 : 0) + (v & 3) * 2 + j;
        int col = kblk * 32 + K;
        float val = emb[tok * HID + col] * scale *
                    imask[(t * BATCH + b) * HID + col];
        dst[id * 2 + j] = (_Float16)val;
    }
}

__global__ __launch_bounds__(256) void prep_x1_kernel(const float* __restrict__ hdrop,
                                                      const float* __restrict__ imask,
                                                      _Float16* __restrict__ dst) {
    int id    = blockIdx.x * 256 + threadIdx.x;
    int v     = id & 7;
    int lane  = (id >> 3) & 31;
    int kblk  = (id >> 8) & 7;
    int mtile = id >> 11;
    int laneN = lane & 15;
    int gA    = lane >> 4;
    int row = mtile * 16 + laneN;
    int t   = row >> 6;
    int b   = row & 63;
#pragma unroll
    for (int j = 0; j < 2; ++j) {
        int K   = gA * 8 + ((v & 4) ? 16 : 0) + (v & 3) * 2 + j;
        int col = kblk * 32 + K;
        float val = hdrop[(t * BATCH + b) * HID + col] * 2.0f *
                    imask[(t * BATCH + b) * HID + col];
        dst[id * 2 + j] = (_Float16)val;
    }
}

// ---------------------------------------------------------------------------
// 3) XW = Asw[16384,256] @ Bsw[256,1024] + bias -> f16 in D-fragment layout:
//    XWsw[(rowTile*64 + ntile)*32 + lane][8 f16]  (one b128 per tile).
//    1 wave per (mtile, 8-ntile group): A-fragment reused 8x, 64 WMMA.
// ---------------------------------------------------------------------------
__global__ __launch_bounds__(256) void gemm_xw_kernel(const _Float16* __restrict__ Asw,
                                                      const _Float16* __restrict__ Bsw,
                                                      const float* __restrict__ bias,
                                                      _Float16* __restrict__ out) {
    int wave  = threadIdx.x >> 5;
    int lane  = threadIdx.x & 31;
    int gw    = blockIdx.x * 8 + wave;      // 0..8191
    int mtile = gw >> 3;                    // 0..1023
    int oct   = gw & 7;                     // ntiles oct*8 .. +7
    int laneN = lane & 15;

    const v16h* A = (const v16h*)Asw;
    const v16h* B = (const v16h*)Bsw;

    v8f acc[8];
#pragma unroll
    for (int n = 0; n < 8; ++n) {
        float bv = bias[(oct * 8 + n) * 16 + laneN];
#pragma unroll
        for (int e = 0; e < 8; ++e) acc[n][e] = bv;
    }
    for (int kblk = 0; kblk < 8; ++kblk) {
        v16h a = A[(mtile * 8 + kblk) * 32 + lane];
#pragma unroll
        for (int n = 0; n < 8; ++n) {
            int ntile = oct * 8 + n;
            v16h b = B[(ntile * 8 + kblk) * 32 + lane];
            acc[n] = wmma_f16(a, b, acc[n]);
        }
    }
    // pack each tile's 8 f32 -> 8 f16, single b128 store per tile
    v8h* outv = (v8h*)out;
#pragma unroll
    for (int n = 0; n < 8; ++n) {
        int tile = mtile * 64 + (oct * 8 + n);
        v8h p;
#pragma unroll
        for (int e = 0; e < 8; ++e) p[e] = (_Float16)acc[n][e];
        outv[tile * 32 + lane] = p;
    }
}

// ---------------------------------------------------------------------------
// 4) Sequential LSTM recurrence. One persistent workgroup (32 waves, 1 WGP).
//    Wave w: mtile = w&3 (16 batch rows), h-tiles {w>>2, (w>>2)+8}.
//    Dynamic LDS: hA (32 KB) + double-buffered XW slab (2 x 128 KB).
// ---------------------------------------------------------------------------
__global__ __launch_bounds__(1024, 1) void lstm_rec_kernel(const _Float16* __restrict__ XWsw,
                                                           const _Float16* __restrict__ Usw,
                                                           const float* __restrict__ omask,
                                                           float* __restrict__ hdrop) {
    extern __shared__ char smem[];
    v16h*     hA   = (v16h*)smem;                       // 1024 frags = 32 KB
    _Float16* xwS0 = (_Float16*)(smem + 32768);         // 128 KB
    _Float16* xwS1 = xwS0 + STEP_HALVES;                // 128 KB

    int tid   = threadIdx.x;
    int wave  = tid >> 5;
    int lane  = tid & 31;
    int laneN = lane & 15;
    int gC    = lane >> 4;
    int mtile = wave & 3;
    int htb   = wave >> 2;              // h-tiles htb and htb+8

    {   // h = 0
        v16h z;
#pragma unroll
        for (int i = 0; i < 16; ++i) z[i] = (_Float16)0.0f;
        hA[tid] = z;
    }
    float cst[2][8];
#pragma unroll
    for (int u = 0; u < 2; ++u)
#pragma unroll
        for (int e = 0; e < 8; ++e) cst[u][e] = 0.0f;

#if HAVE_ASYNC_LDS
    // prologue: stage step 0's XW slab (contiguous 128 KB) into xwS0
    {
        const char* gsrc = (const char*)XWsw;
        char*       ldst = (char*)xwS0;
#pragma unroll
        for (int r = 0; r < 8; ++r) {
            int off = tid * 16 + r * 16384;
            __builtin_amdgcn_global_load_async_to_lds_b128(
                (gv4i_t*)(gsrc + off), (sv4i_t*)(ldst + off), 0, 0);
        }
#if __has_builtin(__builtin_amdgcn_s_wait_asynccnt)
        __builtin_amdgcn_s_wait_asynccnt(0);
#else
        asm volatile("s_wait_asynccnt 0" ::: "memory");
#endif
    }
#endif
    __syncthreads();

    const v16h* U = (const v16h*)Usw;
    _Float16* hA16 = (_Float16*)hA;
    int cur = 0;

    for (int t = 0; t < T_STEPS; ++t) {
        // ---- z init from XW (D-fragment layout: one b128 per tile) ----
        v8f acc[2][4];
#if HAVE_ASYNC_LDS
        const v8h* xwTile = (const v8h*)(cur ? xwS1 : xwS0);
#else
        const v8h* xwTile = (const v8h*)(XWsw + (size_t)t * STEP_HALVES);
#endif
#pragma unroll
        for (int u = 0; u < 2; ++u) {
            int htile = htb + 8 * u;
#pragma unroll
            for (int g = 0; g < 4; ++g) {
                int tile = mtile * 64 + (g * 16 + htile);
                v8h xv = xwTile[tile * 32 + lane];
#pragma unroll
                for (int e = 0; e < 8; ++e) acc[u][g][e] = (float)xv[e];
            }
        }

#if HAVE_ASYNC_LDS
        // ---- kick off async staging of step t+1's slab ----
        if (t + 1 < T_STEPS) {
            const char* gsrc = (const char*)(XWsw + (size_t)(t + 1) * STEP_HALVES);
            char*       ldst = (char*)(cur ? xwS0 : xwS1);
#pragma unroll
            for (int r = 0; r < 8; ++r) {
                int off = tid * 16 + r * 16384;
                __builtin_amdgcn_global_load_async_to_lds_b128(
                    (gv4i_t*)(gsrc + off), (sv4i_t*)(ldst + off), 0, 0);
            }
        }
#else
        if (t + 1 < T_STEPS)
            __builtin_prefetch(XWsw + (size_t)(t + 1) * STEP_HALVES + tid * 16, 0, 1);
#endif

        // ---- z += h @ U (8 K-blocks x 8 WMMA) ----
        for (int kblk = 0; kblk < 8; ++kblk) {
            v16h a = hA[(mtile * 8 + kblk) * 32 + lane];
#pragma unroll
            for (int u = 0; u < 2; ++u) {
                int htile = htb + 8 * u;
#pragma unroll
                for (int g = 0; g < 4; ++g) {
                    int ntile = g * 16 + htile;
                    v16h b = U[(ntile * 8 + kblk) * 32 + lane];
                    acc[u][g] = wmma_f16(a, b, acc[u][g]);
                }
            }
        }

#if HAVE_ASYNC_LDS
#if __has_builtin(__builtin_amdgcn_s_wait_asynccnt)
        __builtin_amdgcn_s_wait_asynccnt(0);
#else
        asm volatile("s_wait_asynccnt 0" ::: "memory");
#endif
#endif
        __syncthreads();   // hA(t-1) reads done; all waves' async writes done

        // ---- gates, cell/hidden update, emit dropped output ----
#pragma unroll
        for (int u = 0; u < 2; ++u) {
            int htile = htb + 8 * u;
#pragma unroll
            for (int e = 0; e < 8; ++e) {
                float iv = acc[u][0][e];
                float jv = acc[u][1][e];
                float fv = acc[u][2][e];
                float ov = acc[u][3][e];
                float cc = sigmoidf_(fv + 1.0f) * cst[u][e] +
                           sigmoidf_(iv) * tanhf(jv);
                cst[u][e] = cc;
                float hv = sigmoidf_(ov) * tanhf(cc);

                int brow = mtile * 16 + e + 8 * gC;     // batch row 0..63
                int col  = htile * 16 + laneN;          // hidden col 0..255
                float omv = omask[(t * BATCH + brow) * HID + col];
                hdrop[(t * BATCH + brow) * HID + col] = hv * omv * 2.0f;

                // scatter h (undropped) into LDS A-fragment layout
                int kblk2 = col >> 5;
                int kin   = col & 31;
                int j  = kin & 1;
                int tt = kin >> 1;
                int v  = ((tt >> 3) << 2) | (tt & 3);
                int gA = (tt >> 2) & 1;
                int lane2 = (brow & 15) + (gA << 4);
                hA16[(((mtile * 8 + kblk2) * 32 + lane2) << 4) + (v << 1) + j] =
                    (_Float16)hv;
            }
        }
        __syncthreads();   // hA(t) visible before next step's WMMA
        cur ^= 1;
    }
}

// ---------------------------------------------------------------------------
// 8) Head: logits[b*T+t, c] = h1drop[t*B+b, :] @ head_w + head_b; pred = >0.
// ---------------------------------------------------------------------------
__global__ __launch_bounds__(256) void head_kernel(const float* __restrict__ hdrop,
                                                   const float* __restrict__ hw,
                                                   const float* __restrict__ hb,
                                                   float* __restrict__ out) {
    int tid = blockIdx.x * 256 + threadIdx.x;   // 0..16383 == t*64+b
    int t = tid >> 6;
    int b = tid & 63;
    float acc[6];
#pragma unroll
    for (int c = 0; c < 6; ++c) acc[c] = hb[c];
    const float* hr = &hdrop[tid * HID];
    for (int h = 0; h < HID; ++h) {
        float v = hr[h];
        const float* wr = &hw[h * 6];
#pragma unroll
        for (int c = 0; c < 6; ++c) acc[c] = fmaf(v, wr[c], acc[c]);
    }
    int orow = b * T_STEPS + t;                 // [B*T, C] row-major
#pragma unroll
    for (int c = 0; c < 6; ++c) {
        float lg = acc[c];
        out[orow * 6 + c]             = lg;
        out[MROWS * 6 + orow * 6 + c] = (lg > 0.0f) ? 1.0f : 0.0f;
    }
}

// ---------------------------------------------------------------------------
// Launcher
// ---------------------------------------------------------------------------
extern "C" void kernel_launch(void* const* d_in, const int* in_sizes, int n_in,
                              void* d_out, int out_size, void* d_ws, size_t ws_size,
                              hipStream_t stream) {
    const int*   tokens = (const int*)  d_in[0];
    const float* emb    = (const float*)d_in[1];
    const float* W0     = (const float*)d_in[2];
    const float* U0     = (const float*)d_in[3];
    const float* b0     = (const float*)d_in[4];
    const float* W1     = (const float*)d_in[5];
    const float* U1     = (const float*)d_in[6];
    const float* b1     = (const float*)d_in[7];
    const float* head_w = (const float*)d_in[8];
    const float* head_b = (const float*)d_in[9];
    const float* imask  = (const float*)d_in[10];   // [2,T,B,H]
    const float* omask  = (const float*)d_in[11];   // [2,T,B,H]
    float* out = (float*)d_out;

    const size_t TBH = (size_t)T_STEPS * BATCH * HID;   // 4,194,304

    // Workspace layout (58 MB total)
    char* ws = (char*)d_ws;
    _Float16* W0sw = (_Float16*)(ws + 0);                      // 512 KB each
    _Float16* U0sw = (_Float16*)(ws + (512u << 10));
    _Float16* W1sw = (_Float16*)(ws + (1024u << 10));
    _Float16* U1sw = (_Float16*)(ws + (1536u << 10));
    _Float16* Xsw  = (_Float16*)(ws + (2u << 20));             // 8 MB
    _Float16* XWsw = (_Float16*)(ws + (10u << 20));            // 32 MB
    float*    hdp  = (float*)   (ws + (42u << 20));            // 16 MB

    const size_t recLds = 32768 + 2 * STEP_HALVES * sizeof(_Float16); // 288 KB

    prep_b_kernel<<<512, 256, 0, stream>>>(W0, W0sw);
    prep_b_kernel<<<512, 256, 0, stream>>>(U0, U0sw);
    prep_b_kernel<<<512, 256, 0, stream>>>(W1, W1sw);
    prep_b_kernel<<<512, 256, 0, stream>>>(U1, U1sw);

    // ---- layer 0 ----
    embed_swz_kernel<<<8192, 256, 0, stream>>>(tokens, emb, imask, Xsw);
    gemm_xw_kernel<<<1024, 256, 0, stream>>>(Xsw, W0sw, b0, XWsw);
    lstm_rec_kernel<<<1, 1024, recLds, stream>>>(XWsw, U0sw, omask, hdp);

    // ---- layer 1 ----
    prep_x1_kernel<<<8192, 256, 0, stream>>>(hdp, imask + TBH, Xsw);
    gemm_xw_kernel<<<1024, 256, 0, stream>>>(Xsw, W1sw, b1, XWsw);
    lstm_rec_kernel<<<1, 1024, recLds, stream>>>(XWsw, U1sw, omask + TBH, hdp);

    // ---- head ----
    head_kernel<<<64, 256, 0, stream>>>(hdp, head_w, head_b, out);

    (void)in_sizes; (void)n_in; (void)out_size; (void)ws_size;
}
